// PointerHead_62758062129878
// MI455X (gfx1250) — matrix operand order repaired and checked by reading, not compiled, not run on hardware
//
#include <hip/hip_runtime.h>
#include <math.h>

typedef float  f32x4  __attribute__((ext_vector_type(4)));
typedef float  v8f    __attribute__((ext_vector_type(8)));
typedef __bf16 v16bf  __attribute__((ext_vector_type(16)));
typedef __bf16 bf16x8 __attribute__((ext_vector_type(8)));

#define DIM   2048
#define NB    8192
#define ROWS  (3*NB)          // 24576
#define BPG   16              // batches per workgroup
#define RPG   48              // rows per workgroup (16 batches * 3)
#define SX    (DIM + 8)       // padded LDS row stride (bf16): 4112B -> conflict-free b128
#define AXS   40              // GEMM1 LDS tile stride (80B)

struct Frag { union { v16bf v; f32x4 q[2]; }; };

__device__ __forceinline__ v8f wmma_bf16(v16bf a, v16bf b, v8f c) {
  // D = A(16x32 bf16) * B(32x16 bf16) + C(f32)
  return __builtin_amdgcn_wmma_f32_16x16x32_bf16(false, a, false, b, (short)0, c, false, false);
}
// B operand: per-lane 16 consecutive bf16 (K = khalf*16 .. +15 at column n)
__device__ __forceinline__ void load_bfrag(Frag& f, const __bf16* p) {
  f.q[0] = *(const f32x4*)p;
  f.q[1] = *(const f32x4*)(p + 8);
}
// A operand: per-lane 16 consecutive bf16 (K = half*16 .. +15 of row m)
__device__ __forceinline__ void load_afrag(Frag& f, const __bf16* p) {
  f.q[0] = *(const f32x4*)p;
  f.q[1] = *(const f32x4*)(p + 8);
}

// ---------------- zero-init bias-vector accumulators ----------------
__global__ void k_zero(float* __restrict__ p, int n) {
  int i = blockIdx.x * 256 + threadIdx.x;
  if (i < n) p[i] = 0.f;
}

// ---------------- u[d]=Wq^T bk, w[d]=Wk^T bq, wsum[d]=colsum(Wv) ----------------
__global__ void k_prep_vec(const float* __restrict__ W, const float* __restrict__ b,
                           float* __restrict__ u, float* __restrict__ w,
                           float* __restrict__ wsum) {
  int col = blockIdx.x * 256 + threadIdx.x;   // grid.x = 8 -> 2048 cols
  int f0  = blockIdx.y * 256;                 // grid.y = 8 -> f chunks
  float au = 0.f, aw = 0.f, as = 0.f;
  for (int f = f0; f < f0 + 256; ++f) {
    au += W[(size_t)f * DIM + col]              * b[DIM + f];
    aw += W[(size_t)(DIM + f) * DIM + col]      * b[f];
    as += W[(size_t)(2 * DIM + f) * DIM + col];
  }
  atomicAdd(u + col, au);
  atomicAdd(w + col, aw);
  atomicAdd(wsum + col, as);
}

// ---------------- c = bq.bk ; bsum = sum(bv) ----------------
__global__ void k_prep_scal(const float* __restrict__ b, float* __restrict__ scal) {
  __shared__ float sc[256], sb[256];
  int t = threadIdx.x;
  float ac = 0.f, ab = 0.f;
  for (int i = t; i < DIM; i += 256) { ac += b[i] * b[DIM + i]; ab += b[2 * DIM + i]; }
  sc[t] = ac; sb[t] = ab;
  __syncthreads();
  for (int s = 128; s > 0; s >>= 1) {
    if (t < s) { sc[t] += sc[t + s]; sb[t] += sb[t + s]; }
    __syncthreads();
  }
  if (t == 0) { scal[0] = sc[0]; scal[1] = sb[0]; }
}

// ---------------- Mb[d][e] = sum_f Wq[f,d]*Wk[f,e]  (bf16 out, WMMA) ----------------
__global__ __launch_bounds__(256) void k_build_M(const float* __restrict__ W,
                                                 __bf16* __restrict__ Mb) {
  __shared__ __align__(16) __bf16 At[64 * AXS];   // [d_local][f_local] transposed stage
  __shared__ __align__(16) __bf16 Bt[64 * AXS];   // [e_local][f_local]
  const int tid  = threadIdx.x;
  const int wv   = tid >> 5, lane = tid & 31;
  const int d0   = blockIdx.x * 64, e0 = blockIdx.y * 64;
  const int rt   = wv & 3, cp = wv >> 2;          // 4 row-tiles x 2 col-tile-pairs
  const int m    = lane & 15, half = lane >> 4;
  v8f acc0 = {}, acc1 = {};

  for (int f0 = 0; f0 < DIM; f0 += 32) {
    __syncthreads();
#pragma unroll
    for (int k = 0; k < 8; ++k) {
      int idx = tid + k * 256;                    // 0..2047
      int fl = idx >> 6, dl = idx & 63;
      At[dl * AXS + fl] = (__bf16)W[(size_t)(f0 + fl) * DIM + d0 + dl];
      Bt[dl * AXS + fl] = (__bf16)W[(size_t)(DIM + f0 + fl) * DIM + e0 + dl];
    }
    __syncthreads();
    Frag a, b0, b1;
    const int arow = rt * 16 + m;
    load_afrag(a, &At[arow * AXS + half * 16]);
    const int br0 = (cp * 2 + 0) * 16 + m;
    const int br1 = (cp * 2 + 1) * 16 + m;
    load_bfrag(b0, &Bt[br0 * AXS + half * 16]);
    load_bfrag(b1, &Bt[br1 * AXS + half * 16]);
    acc0 = wmma_bf16(a.v, b0.v, acc0);
    acc1 = wmma_bf16(a.v, b1.v, acc1);
  }
#pragma unroll
  for (int r = 0; r < 8; ++r) {                   // C layout: row = r + 8*half, col = m
    int dr = d0 + rt * 16 + r + 8 * half;
    Mb[(size_t)dr * DIM + e0 + (cp * 2 + 0) * 16 + m] = (__bf16)acc0[r];
    Mb[(size_t)dr * DIM + e0 + (cp * 2 + 1) * 16 + m] = (__bf16)acc1[r];
  }
}

// ---------------- fused Y=X*M^T + scores + softmax + logits ----------------
__global__ __launch_bounds__(256) void k_main(const float* __restrict__ X,
                                              const __bf16* __restrict__ Mb,
                                              const float* __restrict__ u,
                                              const float* __restrict__ w,
                                              const float* __restrict__ wsum,
                                              const float* __restrict__ scal,
                                              float* __restrict__ out) {
  __shared__ __align__(16) __bf16 Xlds[RPG * SX];     // 48 x 2048 bf16, padded (197376 B)
  __shared__ float ybuf[8][RPG * 16];                 // per-wave y staging (24576 B)
  __shared__ float sacc[BPG * 9];                     // per-batch 3x3 score accum
  __shared__ float rowvec[RPG][3];                    // x.u , w.x , x.wsum per row
  const int tid = threadIdx.x, wv = tid >> 5, lane = tid & 31;
  const int m = lane & 15, half = lane >> 4;
  const size_t rowbase = (size_t)blockIdx.x * RPG;

  // ---- stage 48x2048 X rows as bf16 into LDS (X read from HBM exactly once) ----
  const f32x4* Xg = (const f32x4*)(X + rowbase * DIM);
  for (int i = tid; i < RPG * DIM / 8; i += 256) {
    f32x4 a = Xg[i * 2], b = Xg[i * 2 + 1];
    bf16x8 pk;
    pk[0] = (__bf16)a[0]; pk[1] = (__bf16)a[1]; pk[2] = (__bf16)a[2]; pk[3] = (__bf16)a[3];
    pk[4] = (__bf16)b[0]; pk[5] = (__bf16)b[1]; pk[6] = (__bf16)b[2]; pk[7] = (__bf16)b[3];
    int e   = i * 8;
    int row = e / DIM, col = e % DIM;
    *(bf16x8*)&Xlds[row * SX + col] = pk;
  }
  if (tid < BPG * 9) sacc[tid] = 0.f;
  __syncthreads();

  // ---- per-row bias dots: 144 (row,vec) tasks across 8 waves ----
  for (int task = wv; task < RPG * 3; task += 8) {
    int row = task / 3, vs = task % 3;
    const float* vp = (vs == 0) ? u : (vs == 1) ? w : wsum;
    float acc = 0.f;
    for (int c = lane; c < DIM; c += 32)
      acc += (float)Xlds[row * SX + c] * vp[c];
#pragma unroll
    for (int off = 16; off > 0; off >>= 1) acc += __shfl_xor(acc, off, 32);
    if (lane == 0) rowvec[row][vs] = acc;
  }

  // ---- main WMMA loop: wave owns n-tile pairs {2wv,2wv+1} + 16t; M streams from L2 ----
  float sreg[5] = {0.f, 0.f, 0.f, 0.f, 0.f};
  float* yb = &ybuf[wv][0];
  const __bf16* A0 = &Xlds[(0 * 16 + m) * SX + half * 16];
  const __bf16* A1 = &Xlds[(1 * 16 + m) * SX + half * 16];
  const __bf16* A2 = &Xlds[(2 * 16 + m) * SX + half * 16];

  for (int t = 0; t < 8; ++t) {
    const int nt0 = wv * 2 + t * 16;
    const int nt1 = nt0 + 1;
    const __bf16* B0 = Mb + (size_t)(nt0 * 16 + m) * DIM + half * 16;
    const __bf16* B1 = Mb + (size_t)(nt1 * 16 + m) * DIM + half * 16;
    if (t < 7) {  // warm B rows two iterations ahead (global_prefetch_b8)
      __builtin_prefetch(B0 + 32 * 16 * DIM, 0, 1);
      __builtin_prefetch(B1 + 32 * 16 * DIM, 0, 1);
    }

    v8f c00 = {}, c10 = {}, c20 = {}, c01 = {}, c11 = {}, c21 = {};
    Frag a0A, a1A, a2A, b0A, b1A;     // pipeline set A
    Frag a0B, a1B, a2B, b0B, b1B;     // pipeline set B
    load_bfrag(b0A, B0); load_bfrag(b1A, B1);
    load_afrag(a0A, A0); load_afrag(a1A, A1); load_afrag(a2A, A2);

    for (int e0 = 0; e0 < DIM; e0 += 64) {
      // prefetch set B fragments @ e0+32 (always in range: last = 2016)
      load_bfrag(b0B, B0 + e0 + 32); load_bfrag(b1B, B1 + e0 + 32);
      load_afrag(a0B, A0 + e0 + 32); load_afrag(a1B, A1 + e0 + 32); load_afrag(a2B, A2 + e0 + 32);
      c00 = wmma_bf16(a0A.v, b0A.v, c00);
      c10 = wmma_bf16(a1A.v, b0A.v, c10);
      c20 = wmma_bf16(a2A.v, b0A.v, c20);
      c01 = wmma_bf16(a0A.v, b1A.v, c01);
      c11 = wmma_bf16(a1A.v, b1A.v, c11);
      c21 = wmma_bf16(a2A.v, b1A.v, c21);
      if (e0 + 64 < DIM) {            // prefetch set A fragments @ e0+64
        load_bfrag(b0A, B0 + e0 + 64); load_bfrag(b1A, B1 + e0 + 64);
        load_afrag(a0A, A0 + e0 + 64); load_afrag(a1A, A1 + e0 + 64); load_afrag(a2A, A2 + e0 + 64);
      }
      c00 = wmma_bf16(a0B.v, b0B.v, c00);
      c10 = wmma_bf16(a1B.v, b0B.v, c10);
      c20 = wmma_bf16(a2B.v, b0B.v, c20);
      c01 = wmma_bf16(a0B.v, b1B.v, c01);
      c11 = wmma_bf16(a1B.v, b1B.v, c11);
      c21 = wmma_bf16(a2B.v, b1B.v, c21);
    }

    // ---- per-n-tile cleanup: stage y to per-wave LDS, accumulate 3x3 score partials ----
#pragma unroll
    for (int nn = 0; nn < 2; ++nn) {
      const int nt = nn ? nt1 : nt0;
      v8f x0 = nn ? c01 : c00, x1 = nn ? c11 : c10, x2 = nn ? c21 : c20;
#pragma unroll
      for (int r = 0; r < 8; ++r) {   // C layout: row = r + 8*half (+16*tile), col = m
        yb[( 0 + r + 8 * half) * 16 + m] = x0[r];
        yb[(16 + r + 8 * half) * 16 + m] = x1[r];
        yb[(32 + r + 8 * half) * 16 + m] = x2[r];
      }
#pragma unroll
      for (int k = 0; k < 5; ++k) {
        int combo = lane + 32 * k;
        if (combo < BPG * 9) {
          int g = combo / 9, rr = combo % 9, i = rr / 3, j = rr % 3;
          union { bf16x8 v8[2]; __bf16 h[16]; } xv;
          xv.v8[0] = *(const bf16x8*)&Xlds[(3 * g + i) * SX + nt * 16];
          xv.v8[1] = *(const bf16x8*)&Xlds[(3 * g + i) * SX + nt * 16 + 8];
          const f32x4* yp = (const f32x4*)&yb[(3 * g + j) * 16];
          f32x4 y0 = yp[0], y1 = yp[1], y2 = yp[2], y3 = yp[3];
          float s = 0.f;
#pragma unroll
          for (int c = 0; c < 4; ++c) {
            s += (float)xv.h[c]      * y0[c];
            s += (float)xv.h[4 + c]  * y1[c];
            s += (float)xv.h[8 + c]  * y2[c];
            s += (float)xv.h[12 + c] * y3[c];
          }
          sreg[k] += s;
        }
      }
    }
  }
#pragma unroll
  for (int k = 0; k < 5; ++k) {
    int combo = lane + 32 * k;
    if (combo < BPG * 9) atomicAdd(&sacc[combo], sreg[k]);
  }
  __syncthreads();

  // ---- finalize: scores -> softmax -> logits (48 rows per WG) ----
  if (tid < RPG) {
    int g = tid / 3, i = tid % 3;
    float cc = scal[0], bsum = scal[1];
    const float rs = 0.022097086912079608f;  // 1/sqrt(2048)
    float s[3];
#pragma unroll
    for (int j = 0; j < 3; ++j)
      s[j] = (sacc[g * 9 + i * 3 + j] + rowvec[3 * g + i][0] + rowvec[3 * g + j][1] + cc) * rs;
    float mx = fmaxf(s[0], fmaxf(s[1], s[2]));
    float e0 = __expf(s[0] - mx), e1 = __expf(s[1] - mx), e2 = __expf(s[2] - mx);
    float inv = 1.f / (e0 + e1 + e2);
    float logit = (e0 * (rowvec[3 * g + 0][2] + bsum) +
                   e1 * (rowvec[3 * g + 1][2] + bsum) +
                   e2 * (rowvec[3 * g + 2][2] + bsum)) * inv;
    out[rowbase + tid] = logit;   // (blk*16+g)*3 + i == blk*48 + tid
  }
}

extern "C" void kernel_launch(void* const* d_in, const int* in_sizes, int n_in,
                              void* d_out, int out_size, void* d_ws, size_t ws_size,
                              hipStream_t stream) {
  const float* X = (const float*)d_in[0];   // (8192, 3, 2048) f32
  const float* W = (const float*)d_in[1];   // (6144, 2048)   f32
  const float* b = (const float*)d_in[2];   // (6144,)        f32
  float* out = (float*)d_out;               // (8192, 3)      f32

  char*   wsb  = (char*)d_ws;
  __bf16* Mb   = (__bf16*)wsb;                               // 8,388,608 B
  float*  u    = (float*)(wsb + (size_t)DIM * DIM * 2);      // 2048 f32
  float*  w    = u + DIM;                                    // 2048 f32
  float*  wsum = w + DIM;                                    // 2048 f32
  float*  scal = wsum + DIM;                                 // 2 f32

  k_zero<<<(3 * DIM + 255) / 256, 256, 0, stream>>>(u, 3 * DIM);
  k_prep_vec<<<dim3(8, 8), 256, 0, stream>>>(W, b, u, w, wsum);
  k_prep_scal<<<1, 256, 0, stream>>>(b, scal);
  k_build_M<<<dim3(32, 32), 256, 0, stream>>>(W, Mb);
  k_main<<<ROWS / RPG, 256, 0, stream>>>(X, Mb, u, w, wsum, scal, out);
}
